// RotaryPositionMultiHeadAttention_87935160418777
// MI455X (gfx1250) — matrix operand order repaired and checked by reading, not compiled, not run on hardware
//
#include <hip/hip_runtime.h>

// ---------------------------------------------------------------------------
// RotaryPositionMultiHeadAttention for MI455X (gfx1250, wave32, WMMA)
//   b=8, t=1024, f=1024, h=16, d=64
// All matmuls via v_wmma_f32_16x16x32_f16, software-pipelined so fragment
// loads for step k+1 are in flight while step k's WMMAs issue.
// Attention keeps the whole 64x1024 score strip in LDS (CDNA5: 320KB/WGP).
// ---------------------------------------------------------------------------

typedef __attribute__((ext_vector_type(16))) _Float16 v16h;
typedef __attribute__((ext_vector_type(8)))  _Float16 v8h;
typedef __attribute__((ext_vector_type(8)))  float    v8f;

#define BB 8
#define TT 1024
#define FF 1024
#define NH 16
#define DH 64

// Assemble a 16-half WMMA fragment from two contiguous 8-half runs.
__device__ __forceinline__ v16h load_frag_2x8(const _Float16* p0, const _Float16* p1) {
    v8h lo = *(const v8h*)p0;
    v8h hi = *(const v8h*)p1;
    v16h r;
#pragma unroll
    for (int j = 0; j < 8; ++j) { r[j] = lo[j]; r[j + 8] = hi[j]; }
    return r;
}

__device__ __forceinline__ v8f wmma_f16(v16h a, v16h b, v8f c) {
    return __builtin_amdgcn_wmma_f32_16x16x32_f16(false, a, false, b, (short)0, c,
                                                  false, false);
}

// ---------------------------------------------------------------------------
// f32 -> f16 elementwise convert
// ---------------------------------------------------------------------------
__global__ void cvt_f16_kernel(const float* __restrict__ in, _Float16* __restrict__ out,
                               int n) {
    int i = blockIdx.x * blockDim.x + threadIdx.x;
    if (i < n) out[i] = (_Float16)in[i];
}

// ---------------------------------------------------------------------------
// RoPE for q and k, fused with f16 convert.  One thread per (b,t,h,i<32) pair.
// ---------------------------------------------------------------------------
__global__ void rope_kernel(const float* __restrict__ q, const float* __restrict__ k,
                            const float* __restrict__ cosp, const float* __restrict__ sinp,
                            _Float16* __restrict__ qo, _Float16* __restrict__ ko) {
    unsigned idx = blockIdx.x * blockDim.x + threadIdx.x;  // BB*TT*NH*32 threads
    int i  = idx & 31;
    int hh = (idx >> 5) & (NH - 1);
    int ti = (idx >> 9) & (TT - 1);
    int bi = idx >> 19;
    const float c1 = cosp[ti * DH + i];
    const float s1 = sinp[ti * DH + i];
    const float c2 = cosp[ti * DH + i + 32];
    const float s2 = sinp[ti * DH + i + 32];
    size_t base = ((size_t)(bi * TT + ti)) * FF + hh * DH;
    float q1 = q[base + i], q2 = q[base + i + 32];
    qo[base + i]      = (_Float16)(q1 * c1 - q2 * s1);
    qo[base + i + 32] = (_Float16)(q2 * c2 + q1 * s2);
    float k1 = k[base + i], k2 = k[base + i + 32];
    ko[base + i]      = (_Float16)(k1 * c1 - k2 * s1);
    ko[base + i + 32] = (_Float16)(k2 * c2 + k1 * s2);
}

// ---------------------------------------------------------------------------
// Software-pipelined WMMA GEMM core: acc[4] += IN-strip @ W^T-tile.
// A frag: lane m = row, 16-half run at k + 16*asel. B frag: lane m = col g,
// 16-half run at k + 16*asel.  Next step's fragments load before this
// step's WMMAs so waits are partial.
// ---------------------------------------------------------------------------
__device__ __forceinline__ void gemm_core(const _Float16* __restrict__ arow,
                                          const _Float16* __restrict__ w16,
                                          int col0, int m, int asel, v8f acc[4]) {
    const _Float16* br0 = w16 + (size_t)(col0 + m) * FF + 16 * asel;
    const _Float16* br1 = br0 + (size_t)16 * FF;
    const _Float16* br2 = br0 + (size_t)32 * FF;
    const _Float16* br3 = br0 + (size_t)48 * FF;
    const _Float16* ar  = arow + 16 * asel;

    v16h a  = load_frag_2x8(ar, ar + 8);
    v16h b0 = load_frag_2x8(br0, br0 + 8);
    v16h b1 = load_frag_2x8(br1, br1 + 8);
    v16h b2 = load_frag_2x8(br2, br2 + 8);
    v16h b3 = load_frag_2x8(br3, br3 + 8);

    for (int k0 = 32; k0 < FF; k0 += 32) {
        __builtin_prefetch(ar + k0 + 256, 0, 1);          // global_prefetch_b8
        v16h an  = load_frag_2x8(ar + k0, ar + k0 + 8);
        v16h bn0 = load_frag_2x8(br0 + k0, br0 + k0 + 8);
        v16h bn1 = load_frag_2x8(br1 + k0, br1 + k0 + 8);
        v16h bn2 = load_frag_2x8(br2 + k0, br2 + k0 + 8);
        v16h bn3 = load_frag_2x8(br3 + k0, br3 + k0 + 8);
        acc[0] = wmma_f16(a, b0, acc[0]);
        acc[1] = wmma_f16(a, b1, acc[1]);
        acc[2] = wmma_f16(a, b2, acc[2]);
        acc[3] = wmma_f16(a, b3, acc[3]);
        a = an; b0 = bn0; b1 = bn1; b2 = bn2; b3 = bn3;
    }
    acc[0] = wmma_f16(a, b0, acc[0]);
    acc[1] = wmma_f16(a, b1, acc[1]);
    acc[2] = wmma_f16(a, b2, acc[2]);
    acc[3] = wmma_f16(a, b3, acc[3]);
}

// Epilogue variant 1: write f16 to (b, h, t, d), scaled (for Q/K/V).
__global__ __launch_bounds__(128)
void gemm_qkv_kernel(const _Float16* __restrict__ in, const _Float16* __restrict__ w16,
                     const float* __restrict__ bias, _Float16* __restrict__ dst,
                     float scale) {
    const int lane = threadIdx.x & 31;
    const int wave = threadIdx.x >> 5;            // 0..3
    const int row0 = blockIdx.x * 64 + wave * 16;
    const int col0 = blockIdx.y * 64;
    const int m    = lane & 15;
    const int asel = lane >> 4;

    v8f acc[4] = {};
    gemm_core(in + (size_t)(row0 + m) * FF, w16, col0, m, asel, acc);

#pragma unroll
    for (int j = 0; j < 4; ++j) {
        const int g  = col0 + j * 16 + m;   // C-frag column = lane&15
        const float bv = bias[g];
        const int hh = g >> 6, dd = g & 63;
#pragma unroll
        for (int r = 0; r < 8; ++r) {
            const int rowm = row0 + r + 8 * asel;   // C-frag row = r + 8*(lane>>4)
            const int bi = rowm >> 10, ti = rowm & (TT - 1);
            dst[(((size_t)(bi * NH + hh)) * TT + ti) * DH + dd] =
                (_Float16)((acc[j][r] + bv) * scale);
        }
    }
}

// Epilogue variant 2: f32 row-major to d_out (output projection).
__global__ __launch_bounds__(128)
void gemm_out_kernel(const _Float16* __restrict__ in, const _Float16* __restrict__ w16,
                     const float* __restrict__ bias, float* __restrict__ out) {
    const int lane = threadIdx.x & 31;
    const int wave = threadIdx.x >> 5;
    const int row0 = blockIdx.x * 64 + wave * 16;
    const int col0 = blockIdx.y * 64;
    const int m    = lane & 15;
    const int asel = lane >> 4;

    v8f acc[4] = {};
    gemm_core(in + (size_t)(row0 + m) * FF, w16, col0, m, asel, acc);

#pragma unroll
    for (int j = 0; j < 4; ++j) {
        const int g = col0 + j * 16 + m;
        const float bv = bias[g];
#pragma unroll
        for (int r = 0; r < 8; ++r) {
            const int rowm = row0 + r + 8 * asel;
            out[(size_t)rowm * FF + g] = acc[j][r] + bv;
        }
    }
}

// ---------------------------------------------------------------------------
// Attention: one block per (b*h, 64-query strip). 128 threads = 4 waves.
// LDS: full 64x1024 f16 score strip + double-buffered 32x64 transposed V
// tiles.  1/sqrt(d) folded into Q at projection time.
// ---------------------------------------------------------------------------
#define S_STRIDE 1032      // halves; multiple of 8 keeps 16B-aligned fragments
#define VT_STRIDE 40
#define VT_BUF (64 * VT_STRIDE)

__global__ __launch_bounds__(128)
void attention_kernel(const _Float16* __restrict__ Q, const _Float16* __restrict__ K,
                      const _Float16* __restrict__ V, const unsigned char* __restrict__ mask,
                      _Float16* __restrict__ O /* (b,t,f) f16 */) {
    extern __shared__ _Float16 smem[];
    _Float16* S  = smem;                   // [64][S_STRIDE]
    _Float16* Vt = smem + 64 * S_STRIDE;   // 2 x [64 dd][VT_STRIDE k]

    const int lane = threadIdx.x & 31;
    const int wave = threadIdx.x >> 5;     // 0..3 -> rows 16w..16w+15
    const int bh   = blockIdx.y;           // 0..127
    const int bi   = bh >> 4;
    const int hh   = bh & 15;
    const int q0   = blockIdx.x * 64;
    const int m    = lane & 15;
    const int asel = lane >> 4;

    const size_t base = (size_t)bh * TT * DH;

    // Q fragments for this wave's 16 rows (d = 64 -> two K-steps of 32)
    const _Float16* qrow = Q + base + (size_t)(q0 + wave * 16 + m) * DH;
    v16h qa0 = load_frag_2x8(qrow + 16 * asel,      qrow + 16 * asel + 8);
    v16h qa1 = load_frag_2x8(qrow + 32 + 16 * asel, qrow + 32 + 16 * asel + 8);

    // ---- Phase 1: S = (Q K^T) with mask, into LDS as f16 (pipelined) ----
    {
        const _Float16* kr = K + base + (size_t)m * DH + 16 * asel;
        v16h kb0 = load_frag_2x8(kr,      kr + 8);
        v16h kb1 = load_frag_2x8(kr + 32, kr + 40);
        const unsigned char* mrow =
            mask + (size_t)bi * TT * TT + (size_t)(q0 + wave * 16) * TT + m;
        for (int nt = 0; nt < TT / 16; ++nt) {
            v16h kn0, kn1;
            if (nt + 1 < TT / 16) {
                const _Float16* krn = kr + (size_t)(nt + 1) * 16 * DH;
                kn0 = load_frag_2x8(krn,      krn + 8);
                kn1 = load_frag_2x8(krn + 32, krn + 40);
            }
            v8f c = {};
            c = wmma_f16(qa0, kb0, c);
            c = wmma_f16(qa1, kb1, c);
            const int key = nt * 16 + m;
#pragma unroll
            for (int r = 0; r < 8; ++r) {
                const int ml = r + 8 * asel;
                float v = c[r];
                if (mrow[(size_t)ml * TT + nt * 16]) v = -10000.0f;
                S[(wave * 16 + ml) * S_STRIDE + key] = (_Float16)v;
            }
            kb0 = kn0; kb1 = kn1;
        }
    }
    __syncthreads();

    // ---- Phase 2: row softmax in LDS, vectorized 8-wide, 2 lanes/row ----
    {
        const int row  = threadIdx.x >> 1;
        const int part = threadIdx.x & 1;
        v8h* srow = (v8h*)(S + row * S_STRIDE + part * 512);
        float mx = -3.0e38f;
        for (int c = 0; c < 64; ++c) {
            v8h x = srow[c];
#pragma unroll
            for (int j = 0; j < 8; ++j) mx = fmaxf(mx, (float)x[j]);
        }
        mx = fmaxf(mx, __shfl_xor(mx, 1, 32));
        float sum = 0.0f;
        for (int c = 0; c < 64; ++c) {
            v8h x = srow[c];
            v8h p;
#pragma unroll
            for (int j = 0; j < 8; ++j) {
                float e = __expf((float)x[j] - mx);
                sum += e;
                p[j] = (_Float16)e;
            }
            srow[c] = p;                       // unnormalized exp
        }
        sum += __shfl_xor(sum, 1, 32);
        const float inv = 1.0f / sum;
        for (int c = 0; c < 64; ++c) {
            v8h x = srow[c];
#pragma unroll
            for (int j = 0; j < 8; ++j) x[j] = (_Float16)((float)x[j] * inv);
            srow[c] = x;
        }
    }
    __syncthreads();

    // ---- Phase 3: O = P V, double-buffered transposed V staging in LDS ----
    // stage(): 128 threads x 16 halves = one 32x64 V tile, read as 2x v8h.
    {
        const int e0  = threadIdx.x * 16;      // 0..2047, 16-aligned
        const int kl  = e0 >> 6;               // key-local, constant over the 16
        const int dd0 = e0 & 63;
        const _Float16* vsrc0 = V + base + (size_t)kl * DH + dd0;

        // prologue: stage tile 0 into buffer 0
        {
            v8h x0 = *(const v8h*)vsrc0;
            v8h x1 = *(const v8h*)(vsrc0 + 8);
#pragma unroll
            for (int j = 0; j < 8; ++j) {
                Vt[(dd0 + j)     * VT_STRIDE + kl] = x0[j];
                Vt[(dd0 + 8 + j) * VT_STRIDE + kl] = x1[j];
            }
        }
        __syncthreads();

        v8f oacc[4] = {};
        for (int kt = 0; kt < TT / 32; ++kt) {
            _Float16* cur = Vt + (kt & 1) * VT_BUF;
            if (kt + 1 < TT / 32) {            // stage next tile into other buffer
                _Float16* nxt = Vt + ((kt + 1) & 1) * VT_BUF;
                const _Float16* vsrc = vsrc0 + (size_t)(kt + 1) * 32 * DH;
                v8h x0 = *(const v8h*)vsrc;
                v8h x1 = *(const v8h*)(vsrc + 8);
#pragma unroll
                for (int j = 0; j < 8; ++j) {
                    nxt[(dd0 + j)     * VT_STRIDE + kl] = x0[j];
                    nxt[(dd0 + 8 + j) * VT_STRIDE + kl] = x1[j];
                }
            }
            const _Float16* prow = S + (wave * 16 + m) * S_STRIDE + kt * 32 + 16 * asel;
            v16h pa = load_frag_2x8(prow, prow + 8);
#pragma unroll
            for (int j = 0; j < 4; ++j) {
                const _Float16* vrow = cur + (j * 16 + m) * VT_STRIDE + 16 * asel;
                v16h vb = load_frag_2x8(vrow, vrow + 8);
                oacc[j] = wmma_f16(pa, vb, oacc[j]);
            }
            __syncthreads();                   // one barrier per iteration
        }

        // epilogue: write (b, t, h*64+dd) f16 for the output projection GEMM
#pragma unroll
        for (int j = 0; j < 4; ++j) {
            const int dd = j * 16 + m;
#pragma unroll
            for (int r = 0; r < 8; ++r) {
                const int q = q0 + wave * 16 + r + 8 * asel;
                O[((size_t)(bi * TT + q)) * FF + hh * DH + dd] = (_Float16)oacc[j][r];
            }
        }
    }
}

// ---------------------------------------------------------------------------
// Host-side orchestration
// ---------------------------------------------------------------------------
extern "C" void kernel_launch(void* const* d_in, const int* in_sizes, int n_in,
                              void* d_out, int out_size, void* d_ws, size_t ws_size,
                              hipStream_t stream) {
    const float* q    = (const float*)d_in[0];
    const float* k    = (const float*)d_in[1];
    const float* v    = (const float*)d_in[2];
    const float* cosp = (const float*)d_in[3];
    const float* sinp = (const float*)d_in[4];
    const unsigned char* mask = (const unsigned char*)d_in[5];   // bool array
    const float* Wq = (const float*)d_in[6];
    const float* bq = (const float*)d_in[7];
    const float* Wk = (const float*)d_in[8];
    const float* bk = (const float*)d_in[9];
    const float* Wv = (const float*)d_in[10];
    const float* bv = (const float*)d_in[11];
    const float* Wo = (const float*)d_in[12];
    const float* bo = (const float*)d_in[13];
    float* out = (float*)d_out;

    char* ws = (char*)d_ws;
    const size_t MB = 1024 * 1024;
    _Float16* w16q = (_Float16*)(ws + 0 * MB);     // 2MB each
    _Float16* w16k = (_Float16*)(ws + 2 * MB);
    _Float16* w16v = (_Float16*)(ws + 4 * MB);
    _Float16* w16o = (_Float16*)(ws + 6 * MB);
    _Float16* qin  = (_Float16*)(ws + 8 * MB);     // 16MB each
    _Float16* kin  = (_Float16*)(ws + 24 * MB);
    _Float16* vin  = (_Float16*)(ws + 40 * MB);
    _Float16* Q16  = (_Float16*)(ws + 56 * MB);
    _Float16* K16  = (_Float16*)(ws + 72 * MB);
    _Float16* V16  = (_Float16*)(ws + 88 * MB);
    _Float16* A16  = (_Float16*)(ws + 104 * MB);   // ends at 120MB

    // weight + value f16 conversion
    cvt_f16_kernel<<<(FF * FF) / 256, 256, 0, stream>>>(Wq, w16q, FF * FF);
    cvt_f16_kernel<<<(FF * FF) / 256, 256, 0, stream>>>(Wk, w16k, FF * FF);
    cvt_f16_kernel<<<(FF * FF) / 256, 256, 0, stream>>>(Wv, w16v, FF * FF);
    cvt_f16_kernel<<<(FF * FF) / 256, 256, 0, stream>>>(Wo, w16o, FF * FF);
    cvt_f16_kernel<<<(BB * TT * FF) / 256, 256, 0, stream>>>(v, vin, BB * TT * FF);

    // RoPE + f16 convert for q/k
    rope_kernel<<<(BB * TT * NH * 32) / 256, 256, 0, stream>>>(q, k, cosp, sinp, qin, kin);

    // projections -> (b, h, t, d) f16; 1/sqrt(64) folded into Q
    dim3 ggrid(BB * TT / 64, FF / 64);
    gemm_qkv_kernel<<<ggrid, 128, 0, stream>>>(qin, w16q, bq, Q16, 0.125f);
    gemm_qkv_kernel<<<ggrid, 128, 0, stream>>>(kin, w16k, bk, K16, 1.0f);
    gemm_qkv_kernel<<<ggrid, 128, 0, stream>>>(vin, w16v, bv, V16, 1.0f);

    // attention with whole score strip resident in LDS (CDNA5 320KB/WGP)
    const size_t smem_bytes =
        (size_t)(64 * S_STRIDE + 2 * VT_BUF) * sizeof(_Float16);
    attention_kernel<<<dim3(TT / 64, BB * NH), 128, smem_bytes, stream>>>(
        Q16, K16, V16, mask, A16);

    // output projection -> f32 d_out
    gemm_out_kernel<<<ggrid, 128, 0, stream>>>(A16, w16o, bo, out);
}
